// Transformer_26817775796318
// MI455X (gfx1250) — matrix operand order repaired and checked by reading, not compiled, not run on hardware
//
#include <hip/hip_runtime.h>
#include <hip/hip_bf16.h>

typedef __bf16 bf16;
typedef __attribute__((ext_vector_type(16))) __bf16 v16bf;
typedef __attribute__((ext_vector_type(8)))  __bf16 v8bf;
typedef __attribute__((ext_vector_type(8)))  float  v8f;
typedef __attribute__((ext_vector_type(4)))  int    v4i;

union Frag16 { v16bf v; v8bf h[2]; };

#define BMT 128
#define BNT 128
#define BKT 32
#define LDT 40   // BKT + 8 pad -> 80B rows, 16B aligned

// CDNA5 async global->LDS DMA (ASYNCcnt-tracked, no data VGPRs).
// Probe round 2: builtin exists, signature is
//   (v4i addrspace(1)*, v4i addrspace(3)*, imm int, imm int)
#if defined(__AMDGCN__) && __has_builtin(__builtin_amdgcn_global_load_async_to_lds_b128)
#define USE_ASYNC_LDS 1
#define ASYNC_CP_B128(g, l)                                                   \
  __builtin_amdgcn_global_load_async_to_lds_b128(                             \
      (__attribute__((address_space(1))) v4i*)(g),                            \
      (__attribute__((address_space(3))) v4i*)(l), 0, 0)
#else
#define USE_ASYNC_LDS 0
#endif

#if USE_ASYNC_LDS
static __device__ __forceinline__ void wait_async0() {
#if __has_builtin(__builtin_amdgcn_s_wait_asynccnt)
  __builtin_amdgcn_s_wait_asynccnt(0);
#else
  asm volatile("s_wait_asynccnt 0" ::: "memory");
#endif
}
#endif

// ---------------------------------------------------------------------------
// NT GEMM: C[M,N] = A[M,K](bf16,row-major,lda) * B[N,K](bf16,row-major,ldb)^T
// OUTM: 0 = fp32 out, 1 = bf16 out, 2 = bf16 transposed out (for V^T)
// Per-z slice offsets: off = (z/zdiv)*Sd + (z%zdiv)*Sm  for A, B, C.
// ---------------------------------------------------------------------------
template<int OUTM, bool BIAS, bool RELU>
__global__ __launch_bounds__(256)
void gemm_nt(const bf16* __restrict__ Ag0, const bf16* __restrict__ Bg0,
             const float* __restrict__ bias, void* __restrict__ Cout,
             int K, int lda, int ldb, int ldc, float alpha, int zdiv,
             long aSd, long aSm, long bSd, long bSm, long cSd, long cSm,
             int tT, int tLd)
{
  __shared__ alignas(16) bf16 As[2][BMT * LDT];
  __shared__ alignas(16) bf16 Bs[2][BNT * LDT];

  const int tid  = threadIdx.x;
  const int wave = tid >> 5;
  const int lane = tid & 31;
  const int lh   = lane >> 4;   // half-wave
  const int lm   = lane & 15;
  const int wm   = (wave >> 1) * 32;   // wave M offset (4 rows of waves)
  const int wn   = (wave & 1) * 64;    // wave N offset (2 cols of waves)

  const int  z  = blockIdx.z;
  const long zq = z / zdiv;
  const long zr = z - zq * zdiv;

  const bf16* Ag = Ag0 + zq * aSd + zr * aSm + (long)blockIdx.y * BMT * lda;
  const bf16* Bg = Bg0 + zq * bSd + zr * bSm + (long)blockIdx.x * BNT * ldb;
  const long cBase = zq * cSd + zr * cSm;

  const v8f vzero = {0.f, 0.f, 0.f, 0.f, 0.f, 0.f, 0.f, 0.f};
  v8f acc[2][4];
#pragma unroll
  for (int i = 0; i < 2; ++i)
#pragma unroll
    for (int j = 0; j < 4; ++j) acc[i][j] = vzero;

  const int lrow = tid >> 2;      // 0..63
  const int lch  = tid & 3;       // 16B chunk within 32-elem row
  const int nk = K / BKT;

#if USE_ASYNC_LDS
  // hardware DMA: global -> LDS, no intermediate VGPRs
  {
    ASYNC_CP_B128(Ag + (long)lrow        * lda + lch * 8, &As[0][lrow        * LDT + lch * 8]);
    ASYNC_CP_B128(Ag + (long)(lrow + 64) * lda + lch * 8, &As[0][(lrow + 64) * LDT + lch * 8]);
    ASYNC_CP_B128(Bg + (long)lrow        * ldb + lch * 8, &Bs[0][lrow        * LDT + lch * 8]);
    ASYNC_CP_B128(Bg + (long)(lrow + 64) * ldb + lch * 8, &Bs[0][(lrow + 64) * LDT + lch * 8]);
    wait_async0();
  }
  __syncthreads();
#else
  {
    uint4 ta0 = *(const uint4*)(Ag + (long)lrow        * lda + lch * 8);
    uint4 ta1 = *(const uint4*)(Ag + (long)(lrow + 64) * lda + lch * 8);
    uint4 tb0 = *(const uint4*)(Bg + (long)lrow        * ldb + lch * 8);
    uint4 tb1 = *(const uint4*)(Bg + (long)(lrow + 64) * ldb + lch * 8);
    *(uint4*)&As[0][lrow        * LDT + lch * 8] = ta0;
    *(uint4*)&As[0][(lrow + 64) * LDT + lch * 8] = ta1;
    *(uint4*)&Bs[0][lrow        * LDT + lch * 8] = tb0;
    *(uint4*)&Bs[0][(lrow + 64) * LDT + lch * 8] = tb1;
  }
  __syncthreads();
#endif

  int buf = 0;
  for (int kt = 0; kt < nk; ++kt) {
    const bool more = (kt + 1 < nk);
    if (more) {
      const int k0 = (kt + 1) * BKT;
      const int nb = buf ^ 1;
#if USE_ASYNC_LDS
      // DMA next tile while this wave runs WMMAs below
      ASYNC_CP_B128(Ag + (long)lrow        * lda + k0 + lch * 8, &As[nb][lrow        * LDT + lch * 8]);
      ASYNC_CP_B128(Ag + (long)(lrow + 64) * lda + k0 + lch * 8, &As[nb][(lrow + 64) * LDT + lch * 8]);
      ASYNC_CP_B128(Bg + (long)lrow        * ldb + k0 + lch * 8, &Bs[nb][lrow        * LDT + lch * 8]);
      ASYNC_CP_B128(Bg + (long)(lrow + 64) * ldb + k0 + lch * 8, &Bs[nb][(lrow + 64) * LDT + lch * 8]);
#else
      // short-lifetime load->store (no values held across the WMMA phase)
      uint4 ta0 = *(const uint4*)(Ag + (long)lrow        * lda + k0 + lch * 8);
      uint4 ta1 = *(const uint4*)(Ag + (long)(lrow + 64) * lda + k0 + lch * 8);
      uint4 tb0 = *(const uint4*)(Bg + (long)lrow        * ldb + k0 + lch * 8);
      uint4 tb1 = *(const uint4*)(Bg + (long)(lrow + 64) * ldb + k0 + lch * 8);
      *(uint4*)&As[nb][lrow        * LDT + lch * 8] = ta0;
      *(uint4*)&As[nb][(lrow + 64) * LDT + lch * 8] = ta1;
      *(uint4*)&Bs[nb][lrow        * LDT + lch * 8] = tb0;
      *(uint4*)&Bs[nb][(lrow + 64) * LDT + lch * 8] = tb1;
      if (kt + 2 < nk) {
        __builtin_prefetch(Ag + (long)lrow * lda + (kt + 2) * BKT + lch * 8, 0, 1);
        __builtin_prefetch(Bg + (long)lrow * ldb + (kt + 2) * BKT + lch * 8, 0, 1);
      }
#endif
    }

    // fragment loads from LDS (b128 reads, layouts per CDNA5 ISA 7.12.2)
    Frag16 af[2], bfr[4];
#pragma unroll
    for (int i = 0; i < 2; ++i) {
      const bf16* pA = &As[buf][(wm + i * 16 + lm) * LDT + 8 * lh];
      af[i].h[0] = *(const v8bf*)pA;         // K = 8*lh .. +7
      af[i].h[1] = *(const v8bf*)(pA + 16);  // K = 16+8*lh .. +7
    }
#pragma unroll
    for (int j = 0; j < 4; ++j) {
      const bf16* pB = &Bs[buf][(wn + j * 16 + lm) * LDT + 8 * lh];
      bfr[j].h[0] = *(const v8bf*)pB;        // K = 8*lh .. +7
      bfr[j].h[1] = *(const v8bf*)(pB + 16); // K = 16+8*lh .. +7
    }
#pragma unroll
    for (int i = 0; i < 2; ++i)
#pragma unroll
      for (int j = 0; j < 4; ++j)
        acc[i][j] = __builtin_amdgcn_wmma_f32_16x16x32_bf16(
            false, af[i].v, false, bfr[j].v, (short)0, acc[i][j], false, false);

    if (more) {
#if USE_ASYNC_LDS
      wait_async0();               // __syncthreads does not cover ASYNCcnt
#endif
      __syncthreads();
      buf ^= 1;
    }
  }

  // epilogue (C/D layout: vgpr r -> M = 8*lh + r, lane&15 -> N)
  const int mg0 = blockIdx.y * BMT + wm + lh * 8;
  const int ng0 = blockIdx.x * BNT + wn + lm;
#pragma unroll
  for (int i = 0; i < 2; ++i) {
#pragma unroll
    for (int j = 0; j < 4; ++j) {
      const int ng = ng0 + j * 16;
      float bv = 0.f;
      if (BIAS) bv = bias[ng];
#pragma unroll
      for (int r = 0; r < 8; ++r) {
        const int mg = mg0 + i * 16 + r;
        float vv = acc[i][j][r] * alpha + bv;
        if (RELU) vv = vv > 0.f ? vv : 0.f;
        if (OUTM == 0) {
          ((float*)Cout)[cBase + (long)mg * ldc + ng] = vv;
        } else if (OUTM == 1) {
          ((bf16*)Cout)[cBase + (long)mg * ldc + ng] = (bf16)vv;
        } else {
          const long bb = mg / tT;
          const long tt = mg - bb * tT;
          ((bf16*)Cout)[cBase + (bb * (long)tLd + ng) * (long)tT + tt] = (bf16)vv;
        }
      }
    }
  }
}

// ---------------------------------------------------------------------------
// fp32 [rows,cols] -> bf16 transposed [cols,rows], grid.z = layer
// ---------------------------------------------------------------------------
__global__ __launch_bounds__(256)
void transpose_to_bf16(const float* __restrict__ in, bf16* __restrict__ out,
                       int rows, int cols, long inLS, long outLS)
{
  __shared__ float tile[32][33];
  const long lin  = (long)blockIdx.z * inLS;
  const long lout = (long)blockIdx.z * outLS;
  const int c0 = blockIdx.x * 32, r0 = blockIdx.y * 32;
  const int tx = threadIdx.x & 31, ty = threadIdx.x >> 5;
#pragma unroll
  for (int i = 0; i < 32; i += 8)
    tile[ty + i][tx] = in[lin + (long)(r0 + ty + i) * cols + (c0 + tx)];
  __syncthreads();
#pragma unroll
  for (int i = 0; i < 32; i += 8)
    out[lout + (long)(c0 + ty + i) * rows + (r0 + tx)] = (bf16)tile[tx][ty + i];
}

__global__ void f32_to_bf16(const float* __restrict__ in, bf16* __restrict__ out, long n)
{
  long i = (long)blockIdx.x * blockDim.x + threadIdx.x;
  if (i < n) out[i] = (bf16)in[i];
}

// ---------------------------------------------------------------------------
// y = LayerNorm(a + x) * s + b ; writes fp32 + bf16 copies. One block per row, D=512.
// ---------------------------------------------------------------------------
__global__ __launch_bounds__(256)
void add_ln(const float* __restrict__ a, const float* __restrict__ xr,
            const float* __restrict__ s, const float* __restrict__ b,
            float* __restrict__ outF, bf16* __restrict__ outB)
{
  const long row = blockIdx.x;
  const int tid = threadIdx.x;
  const float* pa = a + row * 512;
  const float* px = xr + row * 512;
  float v0 = pa[tid] + px[tid];
  float v1 = pa[tid + 256] + px[tid + 256];
  float sum = v0 + v1;
  float sq  = v0 * v0 + v1 * v1;
#pragma unroll
  for (int off = 16; off > 0; off >>= 1) {
    sum += __shfl_xor(sum, off, 32);
    sq  += __shfl_xor(sq,  off, 32);
  }
  __shared__ float ssum[8], ssq[8];
  const int wv = tid >> 5, ln = tid & 31;
  if (ln == 0) { ssum[wv] = sum; ssq[wv] = sq; }
  __syncthreads();
  float ts = 0.f, tq = 0.f;
#pragma unroll
  for (int i = 0; i < 8; ++i) { ts += ssum[i]; tq += ssq[i]; }
  const float mean = ts * (1.f / 512.f);
  const float var  = tq * (1.f / 512.f) - mean * mean;
  const float inv  = rsqrtf(var + 1e-5f);
  const float o0 = (v0 - mean) * inv * s[tid] + b[tid];
  const float o1 = (v1 - mean) * inv * s[tid + 256] + b[tid + 256];
  outF[row * 512 + tid]       = o0;
  outF[row * 512 + tid + 256] = o1;
  outB[row * 512 + tid]       = (bf16)o0;
  outB[row * 512 + tid + 256] = (bf16)o1;
}

// ---------------------------------------------------------------------------
// In-place row softmax over 1024 fp32 (the d_out attention-weight region),
// clip to [0,1], also emit bf16 copy for the AV GEMM. One block per row.
// ---------------------------------------------------------------------------
__global__ __launch_bounds__(256)
void softmax_clip(float* __restrict__ w, bf16* __restrict__ wb, float scale)
{
  const long row = blockIdx.x;
  const int tid = threadIdx.x;
  float* p = w + row * 1024;
  float v[4];
  float mx = -3.0e38f;
#pragma unroll
  for (int i = 0; i < 4; ++i) { v[i] = p[tid + i * 256] * scale; mx = fmaxf(mx, v[i]); }
#pragma unroll
  for (int off = 16; off > 0; off >>= 1) mx = fmaxf(mx, __shfl_xor(mx, off, 32));
  __shared__ float smax[8], ssum[8];
  const int wv = tid >> 5, ln = tid & 31;
  if (ln == 0) smax[wv] = mx;
  __syncthreads();
#pragma unroll
  for (int i = 0; i < 8; ++i) mx = fmaxf(mx, smax[i]);
  float sum = 0.f;
#pragma unroll
  for (int i = 0; i < 4; ++i) { v[i] = __expf(v[i] - mx); sum += v[i]; }
#pragma unroll
  for (int off = 16; off > 0; off >>= 1) sum += __shfl_xor(sum, off, 32);
  if (ln == 0) ssum[wv] = sum;
  __syncthreads();
  sum = 0.f;
#pragma unroll
  for (int i = 0; i < 8; ++i) sum += ssum[i];
  const float r = 1.f / sum;
#pragma unroll
  for (int i = 0; i < 4; ++i) {
    float o = v[i] * r;
    o = fminf(fmaxf(o, 0.f), 1.f);
    p[tid + i * 256] = o;
    wb[row * 1024 + tid + i * 256] = (bf16)o;
  }
}

// ---------------------------------------------------------------------------
extern "C" void kernel_launch(void* const* d_in, const int* in_sizes, int n_in,
                              void* d_out, int out_size, void* d_ws, size_t ws_size,
                              hipStream_t stream)
{
  (void)in_sizes; (void)n_in; (void)out_size; (void)ws_size;
  const float* x    = (const float*)d_in[0];
  const float* Wq   = (const float*)d_in[1];
  const float* bq   = (const float*)d_in[2];
  const float* Wk   = (const float*)d_in[3];
  const float* bk   = (const float*)d_in[4];
  const float* Wv   = (const float*)d_in[5];
  const float* bv   = (const float*)d_in[6];
  const float* Wo   = (const float*)d_in[7];
  const float* bo   = (const float*)d_in[8];
  const float* W1   = (const float*)d_in[9];
  const float* b1m  = (const float*)d_in[10];
  const float* W2   = (const float*)d_in[11];
  const float* b2m  = (const float*)d_in[12];
  const float* ln1s = (const float*)d_in[13];
  const float* ln1b = (const float*)d_in[14];
  const float* ln2s = (const float*)d_in[15];
  const float* ln2b = (const float*)d_in[16];

  const long Bn = 4, T = 1024, D = 512, Hh = 8, DH = 512, Pp = 4096, MD = 2048;
  const long Mm  = Bn * T;            // 4096
  const long XE  = Bn * T * D;        // 2,097,152 (final x elems)
  const long WTT = Bn * Hh * T * T;   // 33,554,432 per layer

  // ---- workspace layout (bytes) ----
  char* wsp = (char*)d_ws;
  auto take = [&](size_t bytes) -> char* {
    char* p = wsp; wsp += (bytes + 255) & ~(size_t)255; return p;
  };
  bf16* WqT  = (bf16*)take(4 * D * Pp * 2);
  bf16* WkT  = (bf16*)take(4 * D * Pp * 2);
  bf16* WvT  = (bf16*)take(4 * D * Pp * 2);
  bf16* WoT  = (bf16*)take(4 * Pp * D * 2);
  bf16* W1T  = (bf16*)take(4 * D * MD * 2);
  bf16* W2T  = (bf16*)take(4 * MD * D * 2);
  bf16* x_bf  = (bf16*)take(Mm * D * 2);
  bf16* q_bf  = (bf16*)take(Mm * Pp * 2);
  bf16* k_bf  = (bf16*)take(Mm * Pp * 2);
  bf16* vT_bf = (bf16*)take(Mm * Pp * 2);       // [B][H][DH][T]
  bf16* w_bf  = (bf16*)take(Bn * Hh * T * T * 2);
  bf16* o_bf  = (bf16*)take(Mm * Pp * 2);
  bf16* x1_bf = (bf16*)take(Mm * D * 2);
  bf16* h_bf  = (bf16*)take(Mm * MD * 2);
  float* a_f   = (float*)take(Mm * D * 4);
  float* x1_f  = (float*)take(Mm * D * 4);
  float* z_f   = (float*)take(Mm * D * 4);
  float* xcur  = (float*)take(Mm * D * 4);

  dim3 blk(256);

  // ---- one-time weight transpose+convert (L2-resident afterwards) ----
  transpose_to_bf16<<<dim3(Pp / 32, D / 32, 4), blk, 0, stream>>>(Wq, WqT, (int)D, (int)Pp, D * Pp, D * Pp);
  transpose_to_bf16<<<dim3(Pp / 32, D / 32, 4), blk, 0, stream>>>(Wk, WkT, (int)D, (int)Pp, D * Pp, D * Pp);
  transpose_to_bf16<<<dim3(Pp / 32, D / 32, 4), blk, 0, stream>>>(Wv, WvT, (int)D, (int)Pp, D * Pp, D * Pp);
  transpose_to_bf16<<<dim3(D / 32, Pp / 32, 4), blk, 0, stream>>>(Wo, WoT, (int)Pp, (int)D, Pp * D, Pp * D);
  transpose_to_bf16<<<dim3(MD / 32, D / 32, 4), blk, 0, stream>>>(W1, W1T, (int)D, (int)MD, D * MD, D * MD);
  transpose_to_bf16<<<dim3(D / 32, MD / 32, 4), blk, 0, stream>>>(W2, W2T, (int)MD, (int)D, MD * D, MD * D);
  f32_to_bf16<<<dim3((unsigned)(XE / 256)), blk, 0, stream>>>(x, x_bf, XE);

  const float scale = 0.044194173824159216f;   // 1/sqrt(512)
  const float* xprev = x;
  float* w_base = (float*)d_out + XE;

  for (int l = 0; l < 4; ++l) {
    // Q = x @ Wq + bq   -> bf16 [4096, 4096]
    gemm_nt<1, true, false><<<dim3(Pp / 128, Mm / 128, 1), blk, 0, stream>>>(
        x_bf, WqT + (long)l * D * Pp, bq + (long)l * Pp, q_bf,
        (int)D, (int)D, (int)D, (int)Pp, 1.f, 1, 0, 0, 0, 0, 0, 0, 1, 1);
    // K
    gemm_nt<1, true, false><<<dim3(Pp / 128, Mm / 128, 1), blk, 0, stream>>>(
        x_bf, WkT + (long)l * D * Pp, bk + (long)l * Pp, k_bf,
        (int)D, (int)D, (int)D, (int)Pp, 1.f, 1, 0, 0, 0, 0, 0, 0, 1, 1);
    // V -> transposed bf16 [B][H][DH][T]
    gemm_nt<2, true, false><<<dim3(Pp / 128, Mm / 128, 1), blk, 0, stream>>>(
        x_bf, WvT + (long)l * D * Pp, bv + (long)l * Pp, vT_bf,
        (int)D, (int)D, (int)D, (int)Pp, 1.f, 1, 0, 0, 0, 0, 0, 0, (int)T, (int)Pp);

    // logits = scale * Q @ K^T, written straight into d_out's w region (fp32)
    gemm_nt<0, false, false><<<dim3(T / 128, T / 128, (unsigned)(Bn * Hh)), blk, 0, stream>>>(
        q_bf, k_bf, nullptr, w_base + (long)l * WTT,
        (int)DH, (int)Pp, (int)Pp, (int)T, scale, (int)Hh,
        T * Pp, DH, T * Pp, DH, Hh * T * T, T * T, 1, 1);

    // softmax in place (+ clip) + bf16 copy
    softmax_clip<<<dim3((unsigned)(Bn * Hh * T)), blk, 0, stream>>>(
        w_base + (long)l * WTT, w_bf, 1.f);

    // O = W @ V  -> bf16 o [4096, 4096]
    gemm_nt<1, false, false><<<dim3(DH / 128, T / 128, (unsigned)(Bn * Hh)), blk, 0, stream>>>(
        w_bf, vT_bf, nullptr, o_bf,
        (int)T, (int)T, (int)T, (int)Pp, 1.f, (int)Hh,
        Hh * T * T, T * T, Hh * DH * T, DH * T, T * Pp, DH, 1, 1);

    // a = o @ Wo + bo  -> fp32 [4096, 512]
    gemm_nt<0, true, false><<<dim3(D / 128, Mm / 128, 1), blk, 0, stream>>>(
        o_bf, WoT + (long)l * Pp * D, bo + (long)l * D, a_f,
        (int)Pp, (int)Pp, (int)Pp, (int)D, 1.f, 1, 0, 0, 0, 0, 0, 0, 1, 1);

    // x1 = LN1(a + x)
    add_ln<<<dim3((unsigned)Mm), blk, 0, stream>>>(
        a_f, xprev, ln1s + (long)l * D, ln1b + (long)l * D, x1_f, x1_bf);

    // h = relu(x1 @ W1 + b1) -> bf16 [4096, 2048]
    gemm_nt<1, true, true><<<dim3(MD / 128, Mm / 128, 1), blk, 0, stream>>>(
        x1_bf, W1T + (long)l * D * MD, b1m + (long)l * MD, h_bf,
        (int)D, (int)D, (int)D, (int)MD, 1.f, 1, 0, 0, 0, 0, 0, 0, 1, 1);

    // z = h @ W2 + b2 -> fp32 [4096, 512]
    gemm_nt<0, true, false><<<dim3(D / 128, Mm / 128, 1), blk, 0, stream>>>(
        h_bf, W2T + (long)l * MD * D, b2m + (long)l * D, z_f,
        (int)MD, (int)MD, (int)MD, (int)D, 1.f, 1, 0, 0, 0, 0, 0, 0, 1, 1);

    // x = LN2(z + x1); final layer writes directly into d_out's x region
    float* xo = (l == 3) ? (float*)d_out : xcur;
    add_ln<<<dim3((unsigned)Mm), blk, 0, stream>>>(
        z_f, x1_f, ln2s + (long)l * D, ln2b + (long)l * D, xo, x_bf);
    xprev = xo;
  }
}
